// LinearTransformer_16501264352059
// MI455X (gfx1250) — matrix operand order repaired and checked by reading, not compiled, not run on hardware
//
#include <hip/hip_runtime.h>
#include <hip/hip_bf16.h>

typedef __attribute__((ext_vector_type(16))) __bf16 v16bf;
typedef __attribute__((ext_vector_type(8)))  __bf16 v8bf;
typedef __attribute__((ext_vector_type(8)))  float  v8f;

#define NHEADS 8
#define DHEAD  32
#define CIN    256
#define NTOK   4096
#define BATCH  2
#define GN_EPS 1e-5f
#define ATT_SCALE 0.42044820762685725f  /* 32^-0.25 */

static __device__ __forceinline__ v16bf cat16(v8bf lo, v8bf hi) {
  v16bf r;
#pragma unroll
  for (int e = 0; e < 8; ++e) { r[e] = lo[e]; r[e + 8] = hi[e]; }
  return r;
}

// ---------------------------------------------------------------------------
// Kernel 1: GroupNorm (32 groups over 256 ch), write x_norm transposed
// as bf16 [b][n][c] so GEMM B-fragments are contiguous.
// ---------------------------------------------------------------------------
__global__ __launch_bounds__(256)
void gn_kernel(const float* __restrict__ x, const float* __restrict__ gsc,
               const float* __restrict__ gbi, __bf16* __restrict__ xnt) {
  const int CPG = CIN / 32;                 // 8 channels per group
  int bg = blockIdx.x;                      // 0..63
  int bb = bg >> 5;
  int g  = bg & 31;
  size_t base = ((size_t)bb * CIN + (size_t)g * CPG) * NTOK;
  int tid = threadIdx.x;

  float s = 0.f, s2 = 0.f;
  for (int i = tid; i < CPG * NTOK; i += 256) {
    float v = x[base + i];
    s += v; s2 += v * v;
  }
  __shared__ float red[256], red2[256];
  red[tid] = s; red2[tid] = s2;
  __syncthreads();
  for (int off = 128; off > 0; off >>= 1) {
    if (tid < off) { red[tid] += red[tid + off]; red2[tid] += red2[tid + off]; }
    __syncthreads();
  }
  float inv_n = 1.0f / (float)(CPG * NTOK);
  float mean = red[0] * inv_n;
  float var  = red2[0] * inv_n - mean * mean;
  float rstd = rsqrtf(var + GN_EPS);

  for (int i = tid; i < CPG * NTOK; i += 256) {
    int cl = i >> 12;                       // /4096
    int n  = i & (NTOK - 1);
    int c  = g * CPG + cl;
    float v = (x[base + i] - mean) * rstd * gsc[c] + gbi[c];
    xnt[((size_t)bb * NTOK + n) * CIN + c] = (__bf16)v;
  }
}

// ---------------------------------------------------------------------------
// Kernel 2: convert wq, wk, wv, wo (each 256x256 f32) to bf16, packed.
// ---------------------------------------------------------------------------
__global__ __launch_bounds__(256)
void wconv_kernel(const float* __restrict__ wq, const float* __restrict__ wk,
                  const float* __restrict__ wv, const float* __restrict__ wo,
                  __bf16* __restrict__ dst) {
  int i = blockIdx.x * 256 + threadIdx.x;   // 0 .. 262143
  int m = i >> 16;
  int e = i & 65535;
  const float* src = (m == 0) ? wq : (m == 1) ? wk : (m == 2) ? wv : wo;
  dst[(size_t)m * 65536 + e] = (__bf16)src[e];
}

// ---------------------------------------------------------------------------
// Kernel 3: QKV projection GEMM (bf16 WMMA, f32 acc).
//   q/k -> [b][h][n][32] bf16 (SCALE folded in);  v -> [b][h][d][n] bf16.
// One wave per 16x16 output tile; K=256 in 8 WMMAs.
// ---------------------------------------------------------------------------
__global__ __launch_bounds__(128)
void qkv_kernel(const __bf16* __restrict__ xnt, const __bf16* __restrict__ wbf,
                const float* __restrict__ bq, const float* __restrict__ bk,
                const float* __restrict__ bv,
                __bf16* __restrict__ qT, __bf16* __restrict__ kT,
                __bf16* __restrict__ vm) {
  int wid  = threadIdx.x >> 5;
  int lane = threadIdx.x & 31;
  int half = lane >> 4, lanelo = lane & 15;
  int t   = blockIdx.x * 4 + wid;           // 24576 tiles
  int nt  = t & 255;
  int mt  = (t >> 8) & 15;
  int bb  = (t >> 12) & 1;
  int mat = t >> 13;                        // 0=q 1=k 2=v
  int mbase = mt * 16, nbase = nt * 16;

  const __bf16* w    = wbf + (size_t)mat * 65536;
  const float*  bias = (mat == 0) ? bq : (mat == 1) ? bk : bv;
  const __bf16* arow = w + (size_t)(mbase + lanelo) * CIN;
  const __bf16* brow = xnt + ((size_t)bb * NTOK + (nbase + lanelo)) * CIN;

  v8f acc = {};
#pragma unroll
  for (int kb = 0; kb < CIN; kb += 32) {
    v8bf a0 = *(const v8bf*)(arow + kb + half * 8);
    v8bf a1 = *(const v8bf*)(arow + kb + 16 + half * 8);
    v16bf a  = cat16(a0, a1);
    v16bf bf = cat16(*(const v8bf*)(brow + kb + half * 8),
                     *(const v8bf*)(brow + kb + 16 + half * 8));
    acc = __builtin_amdgcn_wmma_f32_16x16x32_bf16(false, a, false, bf,
                                                  (short)0, acc, false, false);
  }

  int n = nbase + lanelo;
#pragma unroll
  for (int r = 0; r < 8; ++r) {
    int o = mbase + r + 8 * half;
    float val = acc[r] + bias[o];
    if (mat == 2) {
      vm[((size_t)bb * CIN + o) * NTOK + n] = (__bf16)val;
    } else {
      int h = o >> 5, d = o & 31;
      __bf16* dst = (mat == 0) ? qT : kT;
      dst[(((size_t)bb * NHEADS + h) * NTOK + n) * DHEAD + d] =
          (__bf16)(val * ATT_SCALE);
    }
  }
}

// ---------------------------------------------------------------------------
// Kernel 4: flash attention. One wave per (b, h, 16-query tile).
// S = Q^T K (one WMMA per 16 keys, K-dim = d = 32), online softmax with
// cross-lane butterflies, P re-striped through LDS, O^T += P V^T (2 WMMAs).
// ---------------------------------------------------------------------------
__global__ __launch_bounds__(128)
void attn_kernel(const __bf16* __restrict__ qT, const __bf16* __restrict__ kT,
                 const __bf16* __restrict__ vm, __bf16* __restrict__ attn) {
  __shared__ __bf16 plds[4][16 * 32];       // per-wave P tile (16 x 32 bf16)
  int wid  = threadIdx.x >> 5;
  int lane = threadIdx.x & 31;
  int half = lane >> 4, lanelo = lane & 15;
  int t  = blockIdx.x * 4 + wid;            // 4096 tiles
  int qt = t & 255;
  int h  = (t >> 8) & 7;
  int bb = t >> 11;
  int qbase = qt * 16;
  size_t bh = (size_t)bb * NHEADS + h;

  // A fragment: Q^T tile (M = 16 queries, K = d = 32), loaded once.
  const __bf16* qrow = qT + (bh * NTOK + qbase + lanelo) * DHEAD;
  v16bf aq = cat16(*(const v8bf*)(qrow + half * 8),
                   *(const v8bf*)(qrow + 16 + half * 8));

  const __bf16* kbase  = kT + bh * NTOK * DHEAD;
  const __bf16* vbase0 = vm + ((size_t)bb * CIN + h * DHEAD + lanelo) * NTOK;
  const __bf16* vbase1 = vm + ((size_t)bb * CIN + h * DHEAD + 16 + lanelo) * NTOK;
  __bf16* myp = plds[wid];

  float m[8], l[8];
  v8f o0 = {}, o1 = {};
#pragma unroll
  for (int r = 0; r < 8; ++r) { m[r] = -1e30f; l[r] = 0.f; }

  for (int jb = 0; jb < NTOK; jb += 32) {
    if (jb + 32 < NTOK)
      __builtin_prefetch(kbase + (size_t)(jb + 32) * DHEAD, 0, 0);

    // B fragments: K tiles (K-dim = d, N = 16 keys each)
    v16bf bk0 = cat16(*(const v8bf*)(kbase + (size_t)(jb + lanelo) * DHEAD + half * 8),
                      *(const v8bf*)(kbase + (size_t)(jb + lanelo) * DHEAD + 16 + half * 8));
    v16bf bk1 = cat16(*(const v8bf*)(kbase + (size_t)(jb + 16 + lanelo) * DHEAD + half * 8),
                      *(const v8bf*)(kbase + (size_t)(jb + 16 + lanelo) * DHEAD + 16 + half * 8));
    v8f z = {};
    v8f s0 = __builtin_amdgcn_wmma_f32_16x16x32_bf16(false, aq, false, bk0,
                                                     (short)0, z, false, false);
    v8f s1 = __builtin_amdgcn_wmma_f32_16x16x32_bf16(false, aq, false, bk1,
                                                     (short)0, z, false, false);
    // Online softmax. C-layout: element r = row (r + 8*half), col = lanelo.
#pragma unroll
    for (int r = 0; r < 8; ++r) {
      float mx = fmaxf(s0[r], s1[r]);
      mx = fmaxf(mx, __shfl_xor(mx, 1, 32));
      mx = fmaxf(mx, __shfl_xor(mx, 2, 32));
      mx = fmaxf(mx, __shfl_xor(mx, 4, 32));
      mx = fmaxf(mx, __shfl_xor(mx, 8, 32));
      float mn   = fmaxf(m[r], mx);
      float corr = __expf(m[r] - mn);
      float p0 = __expf(s0[r] - mn);
      float p1 = __expf(s1[r] - mn);
      float sm = p0 + p1;
      sm += __shfl_xor(sm, 1, 32);
      sm += __shfl_xor(sm, 2, 32);
      sm += __shfl_xor(sm, 4, 32);
      sm += __shfl_xor(sm, 8, 32);
      l[r] = l[r] * corr + sm;
      m[r] = mn;
      o0[r] *= corr;
      o1[r] *= corr;
      int row = r + 8 * half;
      myp[row * 32 + lanelo]      = (__bf16)p0;   // keys jb..jb+15
      myp[row * 32 + 16 + lanelo] = (__bf16)p1;   // keys jb+16..jb+31
    }
    // Re-stripe P (C layout) -> A layout via LDS (same-wave DS ops are in order)
    const __bf16* prow = myp + lanelo * 32;
    v16bf ap = cat16(*(const v8bf*)(prow + half * 8),
                     *(const v8bf*)(prow + 16 + half * 8));
    // B fragments: V^T chunks (K-dim = 32 keys, N = 16 channels each)
    v16bf bv0 = cat16(*(const v8bf*)(vbase0 + jb + half * 8),
                      *(const v8bf*)(vbase0 + jb + 16 + half * 8));
    v16bf bv1 = cat16(*(const v8bf*)(vbase1 + jb + half * 8),
                      *(const v8bf*)(vbase1 + jb + 16 + half * 8));
    o0 = __builtin_amdgcn_wmma_f32_16x16x32_bf16(false, ap, false, bv0,
                                                 (short)0, o0, false, false);
    o1 = __builtin_amdgcn_wmma_f32_16x16x32_bf16(false, ap, false, bv1,
                                                 (short)0, o1, false, false);
  }

  // Epilogue: normalize rows by l, write attn output as [b][n][c] bf16.
#pragma unroll
  for (int r = 0; r < 8; ++r) {
    float inv = 1.0f / l[r];
    int n = qbase + r + 8 * half;
    size_t rowb = ((size_t)bb * NTOK + n) * CIN + h * DHEAD;
    attn[rowb + lanelo]      = (__bf16)(o0[r] * inv);
    attn[rowb + 16 + lanelo] = (__bf16)(o1[r] * inv);
  }
}

// ---------------------------------------------------------------------------
// Kernel 5: output projection GEMM + bias + residual, f32 output.
// ---------------------------------------------------------------------------
__global__ __launch_bounds__(128)
void proj_kernel(const __bf16* __restrict__ attn, const __bf16* __restrict__ wbf,
                 const float* __restrict__ bo, const float* __restrict__ x,
                 float* __restrict__ out) {
  int wid  = threadIdx.x >> 5;
  int lane = threadIdx.x & 31;
  int half = lane >> 4, lanelo = lane & 15;
  int t  = blockIdx.x * 4 + wid;            // 8192 tiles
  int nt = t & 255;
  int mt = (t >> 8) & 15;
  int bb = t >> 12;
  int mbase = mt * 16, nbase = nt * 16;

  const __bf16* w    = wbf + (size_t)3 * 65536;   // wo
  const __bf16* arow = w + (size_t)(mbase + lanelo) * CIN;
  const __bf16* brow = attn + ((size_t)bb * NTOK + (nbase + lanelo)) * CIN;

  v8f acc = {};
#pragma unroll
  for (int kb = 0; kb < CIN; kb += 32) {
    v8bf a0 = *(const v8bf*)(arow + kb + half * 8);
    v8bf a1 = *(const v8bf*)(arow + kb + 16 + half * 8);
    v16bf a  = cat16(a0, a1);
    v16bf bf = cat16(*(const v8bf*)(brow + kb + half * 8),
                     *(const v8bf*)(brow + kb + 16 + half * 8));
    acc = __builtin_amdgcn_wmma_f32_16x16x32_bf16(false, a, false, bf,
                                                  (short)0, acc, false, false);
  }

  int n = nbase + lanelo;
#pragma unroll
  for (int r = 0; r < 8; ++r) {
    int o = mbase + r + 8 * half;
    size_t idx = ((size_t)bb * CIN + o) * NTOK + n;
    out[idx] = acc[r] + bo[o] + x[idx];
  }
}

// ---------------------------------------------------------------------------
// Host launcher
// ---------------------------------------------------------------------------
extern "C" void kernel_launch(void* const* d_in, const int* in_sizes, int n_in,
                              void* d_out, int out_size, void* d_ws, size_t ws_size,
                              hipStream_t stream) {
  const float* x   = (const float*)d_in[0];
  const float* gsc = (const float*)d_in[1];
  const float* gbi = (const float*)d_in[2];
  const float* wq  = (const float*)d_in[3];
  const float* bq  = (const float*)d_in[4];
  const float* wk  = (const float*)d_in[5];
  const float* bk  = (const float*)d_in[6];
  const float* wv  = (const float*)d_in[7];
  const float* bv  = (const float*)d_in[8];
  const float* wo  = (const float*)d_in[9];
  const float* bo  = (const float*)d_in[10];

  char* ws = (char*)d_ws;
  const size_t SZ_XN = (size_t)BATCH * CIN * NTOK * sizeof(__bf16);      // 4 MiB
  const size_t SZ_W  = (size_t)4 * 65536 * sizeof(__bf16);               // 512 KiB
  const size_t SZ_T  = (size_t)BATCH * NHEADS * NTOK * DHEAD * sizeof(__bf16); // 4 MiB

  __bf16* xnt = (__bf16*)(ws);
  __bf16* wbf = (__bf16*)(ws + SZ_XN);
  __bf16* qT  = (__bf16*)(ws + SZ_XN + SZ_W);
  __bf16* kT  = (__bf16*)(ws + SZ_XN + SZ_W + SZ_T);
  __bf16* vm  = (__bf16*)(ws + SZ_XN + SZ_W + 2 * SZ_T);
  __bf16* at  = (__bf16*)(ws + SZ_XN + SZ_W + 3 * SZ_T);
  float*  out = (float*)d_out;

  gn_kernel   <<<BATCH * 32, 256, 0, stream>>>(x, gsc, gbi, xnt);
  wconv_kernel<<<1024, 256, 0, stream>>>(wq, wk, wv, wo, wbf);
  qkv_kernel  <<<6144, 128, 0, stream>>>(xnt, wbf, bq, bk, bv, qT, kT, vm);
  attn_kernel <<<1024, 128, 0, stream>>>(qT, kT, vm, at);
  proj_kernel <<<2048, 128, 0, stream>>>(at, wbf, bo, x, out);
}
